// SelfAttention_1322849927509
// MI455X (gfx1250) — matrix operand order repaired and checked by reading, not compiled, not run on hardware
//
#include <hip/hip_runtime.h>
#include <hip/hip_bf16.h>

typedef __attribute__((ext_vector_type(16))) _Float16 v16h;
typedef __attribute__((ext_vector_type(8)))  _Float16 v8h;
typedef __attribute__((ext_vector_type(8)))  float    v8f;
typedef __attribute__((ext_vector_type(4)))  float    v4f;

#define C_DIM   128
#define N_DIM   4096
#define HEADS   4
#define DHEAD   32
#define HSTRIDE (N_DIM * DHEAD)   // 131072 halves per head for q/k/vT
#define NTOT    (C_DIM * N_DIM)   // 524288
#define WSZ     (C_DIM * C_DIM)   // 16384 elems per weight matrix

// ---------------------------------------------------------------------------
// WMMA helpers (CDNA5 wave32, V_WMMA_F32_16X16X32_F16)
// ---------------------------------------------------------------------------
__device__ __forceinline__ v8f wmma16(v16h a, v16h b, v8f c) {
  // 8 args: (neg_a, A, neg_b, B, c_mod, C, reuse_a, reuse_b)
  return __builtin_amdgcn_wmma_f32_16x16x32_f16(false, a, false, b,
                                                (short)0, c, false, false);
}

// Build a 16-half fragment from two 8-half (16B aligned) segments.
__device__ __forceinline__ v16h frag16(const _Float16* p0, const _Float16* p1) {
  v8h x0 = *reinterpret_cast<const v8h*>(p0);
  v8h x1 = *reinterpret_cast<const v8h*>(p1);
  v16h a;
#pragma unroll
  for (int j = 0; j < 8; ++j) { a[j] = x0[j]; a[8 + j] = x1[j]; }
  return a;
}

// A-matrix fragment (16x32) from an f16 weight row (length 128).
// A layout (matches B): half j -> K = hi*16 + j  => two contiguous
// 8-half segments at c0+hi*16 and c0+hi*16+8.
__device__ __forceinline__ v16h a_frag_h16row(const _Float16* row, int c0, int hi) {
  return frag16(row + c0 + hi * 16, row + c0 + 8 + hi * 16);
}

// ---------------------------------------------------------------------------
// Kernel 0: one-time fp32 -> f16 conversion of the four weight matrices.
// ---------------------------------------------------------------------------
__global__ __launch_bounds__(256) void wconv_kernel(
    const float* __restrict__ qw, const float* __restrict__ kw,
    const float* __restrict__ vw, const float* __restrict__ pw,
    _Float16* __restrict__ w16) {
  const float* src[4] = {qw, kw, vw, pw};
  const float* s = src[blockIdx.y];
  _Float16* d = w16 + blockIdx.y * WSZ;
  const int i = (blockIdx.x * 256 + threadIdx.x) * 4;
  v4f v = *reinterpret_cast<const v4f*>(s + i);
#pragma unroll
  for (int j = 0; j < 4; ++j) d[i + j] = (_Float16)v[j];
}

// ---------------------------------------------------------------------------
// Kernel 1: GroupNorm(num_groups=1) statistics (deterministic single-WG).
// ---------------------------------------------------------------------------
__global__ __launch_bounds__(1024) void gn_stats_kernel(
    const float* __restrict__ x, float* __restrict__ stats) {
  __shared__ float s_sum[1024];
  __shared__ float s_sq[1024];
  const int t = threadIdx.x;
  float sum = 0.f, sq = 0.f;
  for (int i = t * 4; i < NTOT; i += 1024 * 4) {
    v4f v = *reinterpret_cast<const v4f*>(x + i);
#pragma unroll
    for (int j = 0; j < 4; ++j) { sum += v[j]; sq += v[j] * v[j]; }
  }
  s_sum[t] = sum; s_sq[t] = sq;
  __syncthreads();
  for (int off = 512; off > 0; off >>= 1) {
    if (t < off) { s_sum[t] += s_sum[t + off]; s_sq[t] += s_sq[t + off]; }
    __syncthreads();
  }
  if (t == 0) {
    const float inv_n = 1.0f / (float)NTOT;
    float mean = s_sum[0] * inv_n;
    float var  = s_sq[0] * inv_n - mean * mean;
    stats[0] = mean;
    stats[1] = rsqrtf(var + 1e-5f);
  }
}

// ---------------------------------------------------------------------------
// Kernel 2: fused normalize + Q/K/V GEMM.  q,k: [head][n][32]; v: [head][d][N].
// Loop nest: B fragments (shared by q/k/v) loaded ONCE per n-tile; all 12
// A fragments hoisted (loop-invariant).  Q/K epilogue packs the 8 consecutive
// d values per lane into one v8h => single 16B store.
// ---------------------------------------------------------------------------
__global__ __launch_bounds__(256) void qkv_kernel(
    const float* __restrict__ x,  const float* __restrict__ gnw,
    const float* __restrict__ gnb,
    const _Float16* __restrict__ w16,
    const float* __restrict__ qb, const float* __restrict__ kb,
    const float* __restrict__ vb,
    const float* __restrict__ stats,
    _Float16* __restrict__ qo, _Float16* __restrict__ ko,
    _Float16* __restrict__ vto) {
  __shared__ _Float16 hx[64][136];   // [col][c], padded vs bank clash
  const int n0 = blockIdx.x * 64;
  const float mean = stats[0], rstd = stats[1];

  for (int idx = threadIdx.x; idx < C_DIM * 64; idx += 256) {
    const int c = idx >> 6, col = idx & 63;
    const float v = x[c * N_DIM + n0 + col];
    hx[col][c] = (_Float16)(((v - mean) * rstd) * gnw[c] + gnb[c]);
  }
  __syncthreads();

  const int lane = threadIdx.x & 31, w = threadIdx.x >> 5;
  const int l = lane & 15, hi = lane >> 4;
  const int head  = w >> 1;                  // o = 16w..16w+15 => head fixed
  const int dbase = (w & 1) * 16 + hi * 8;   // 8 consecutive d per lane
  const float* BS[3] = {qb, kb, vb};

  // All A fragments (3 matrices x 4 K-chunks) and bias vectors, hoisted.
  v16h A[3][4];
  v4f  bv0[3], bv1[3];
#pragma unroll
  for (int m = 0; m < 3; ++m) {
    const _Float16* wrow = w16 + m * WSZ + (w * 16 + l) * C_DIM;
#pragma unroll
    for (int kc = 0; kc < 4; ++kc) A[m][kc] = a_frag_h16row(wrow, kc * 32, hi);
    const float* bp = BS[m] + head * DHEAD + dbase;
    bv0[m] = *reinterpret_cast<const v4f*>(bp);
    bv1[m] = *reinterpret_cast<const v4f*>(bp + 4);
  }

#pragma unroll
  for (int nt = 0; nt < 4; ++nt) {
    const _Float16* cb = &hx[nt * 16 + l][0];
    // B fragments shared by all three matrices: load once.
    v16h B[4];
#pragma unroll
    for (int kc = 0; kc < 4; ++kc)
      B[kc] = frag16(cb + kc * 32 + hi * 16, cb + kc * 32 + 8 + hi * 16);
    const int n = n0 + nt * 16 + l;
#pragma unroll
    for (int m = 0; m < 3; ++m) {
      v8f acc = {};
      acc = wmma16(A[m][0], B[0], acc);
      acc = wmma16(A[m][1], B[1], acc);
      acc = wmma16(A[m][2], B[2], acc);
      acc = wmma16(A[m][3], B[3], acc);
      if (m < 2) {
        v8h pk;
#pragma unroll
        for (int r = 0; r < 4; ++r) pk[r]     = (_Float16)(acc[r]     + bv0[m][r]);
#pragma unroll
        for (int r = 0; r < 4; ++r) pk[4 + r] = (_Float16)(acc[4 + r] + bv1[m][r]);
        _Float16* dst = (m == 0 ? qo : ko) + head * HSTRIDE + n * DHEAD + dbase;
        *reinterpret_cast<v8h*>(dst) = pk;          // one global_store_b128
      } else {
#pragma unroll
        for (int r = 0; r < 8; ++r) {               // vT scatter: d-major rows
          const int di = dbase + r;
          const float bias = r < 4 ? bv0[m][r] : bv1[m][r - 4];
          vto[head * HSTRIDE + di * N_DIM + n] = (_Float16)(acc[r] + bias);
        }
      }
    }
  }
}

// ---------------------------------------------------------------------------
// Kernel 3: flash attention, 64-key chunks (8 WMMA per chunk). kb-loop is
// unrolled x2 so the scheduler can co-execute one chunk's softmax VALU with
// the next chunk's K loads / S-WMMAs (16-bit WMMA co-executes with VALU).
// Cross-chunk pbuf reuse is safe: LDS ops are wave-in-order + dscnt fence.
// ---------------------------------------------------------------------------
__global__ __launch_bounds__(256) void attn_kernel(
    const _Float16* __restrict__ qm, const _Float16* __restrict__ km,
    const _Float16* __restrict__ vt, _Float16* __restrict__ ao) {
  __shared__ _Float16 pbuf[8][16][64];   // per-wave P tile [query][key]
  const int lane = threadIdx.x & 31, w = threadIdx.x >> 5;
  const int l = lane & 15, hi = lane >> 4;
  const int gw = blockIdx.x * 8 + w;           // 1024 query tiles total
  const int head = gw >> 8, qt = gw & 255;
  const int qn0 = qt * 16;
  const _Float16* qh = qm + head * HSTRIDE;
  const _Float16* kh = km + head * HSTRIDE;
  const _Float16* vh = vt + head * HSTRIDE;
  const float scale = 0.17677669529663687f;    // 1/sqrt(32)

  const _Float16* qrow = qh + (qn0 + l) * DHEAD;
  const v16h aq = frag16(qrow + hi * 16, qrow + 8 + hi * 16);

  float mrow[8], srow[8];
#pragma unroll
  for (int r = 0; r < 8; ++r) { mrow[r] = -1e30f; srow[r] = 0.f; }
  v8f o0 = {}, o1 = {};

#pragma unroll 2
  for (int kb = 0; kb < N_DIM; kb += 64) {
    v8f s[4];
#pragma unroll
    for (int kc = 0; kc < 4; ++kc) {
      const _Float16* kp = kh + (kb + kc * 16 + l) * DHEAD + hi * 16;
      s[kc] = wmma16(aq, frag16(kp, kp + 8), (v8f){});
    }

#pragma unroll
    for (int r = 0; r < 8; ++r) {
      float t = fmaxf(fmaxf(s[0][r], s[1][r]), fmaxf(s[2][r], s[3][r])) * scale;
      t = fmaxf(t, __shfl_xor(t, 1, 32));
      t = fmaxf(t, __shfl_xor(t, 2, 32));
      t = fmaxf(t, __shfl_xor(t, 4, 32));
      t = fmaxf(t, __shfl_xor(t, 8, 32));
      const float mnew = fmaxf(mrow[r], t);
      float p[4], rs = 0.f;
#pragma unroll
      for (int kc = 0; kc < 4; ++kc) {
        p[kc] = __expf(s[kc][r] * scale - mnew);
        rs += p[kc];
      }
      rs += __shfl_xor(rs, 1, 32);
      rs += __shfl_xor(rs, 2, 32);
      rs += __shfl_xor(rs, 4, 32);
      rs += __shfl_xor(rs, 8, 32);
      const float alpha = __expf(mrow[r] - mnew);
      srow[r] = srow[r] * alpha + rs;
      mrow[r] = mnew;
      o0[r] *= alpha;
      o1[r] *= alpha;
      const int q = r + hi * 8;
#pragma unroll
      for (int kc = 0; kc < 4; ++kc)
        pbuf[w][q][kc * 16 + l] = (_Float16)p[kc];
    }
    asm volatile("s_wait_dscnt 0" ::: "memory");   // LDS transpose fence

    const v16h ap0 = frag16(&pbuf[w][l][hi * 16],      &pbuf[w][l][8 + hi * 16]);
    const v16h ap1 = frag16(&pbuf[w][l][32 + hi * 16], &pbuf[w][l][40 + hi * 16]);

    const _Float16* vr0 = vh + l * N_DIM + kb + hi * 16;          // d slice 0
    const _Float16* vr1 = vh + (16 + l) * N_DIM + kb + hi * 16;   // d slice 1
    o0 = wmma16(ap0, frag16(vr0, vr0 + 8), o0);
    o0 = wmma16(ap1, frag16(vr0 + 32, vr0 + 40), o0);
    o1 = wmma16(ap0, frag16(vr1, vr1 + 8), o1);
    o1 = wmma16(ap1, frag16(vr1 + 32, vr1 + 40), o1);
  }

#pragma unroll
  for (int r = 0; r < 8; ++r) {
    const float inv = 1.0f / srow[r];
    const int n = qn0 + r + hi * 8;
    ao[n * C_DIM + head * DHEAD + l]      = (_Float16)(o0[r] * inv);
    ao[n * C_DIM + head * DHEAD + 16 + l] = (_Float16)(o1[r] * inv);
  }
}

// ---------------------------------------------------------------------------
// Kernel 4: projection GEMM + bias + residual (fp32 out).
// ---------------------------------------------------------------------------
__global__ __launch_bounds__(256) void proj_kernel(
    const float* __restrict__ x, const _Float16* __restrict__ pw16,
    const float* __restrict__ pb, const _Float16* __restrict__ ao,
    float* __restrict__ out) {
  const int lane = threadIdx.x & 31, w = threadIdx.x >> 5;
  const int l = lane & 15, hi = lane >> 4;
  const int n0 = blockIdx.x * 64;
  const _Float16* wrow = pw16 + (w * 16 + l) * C_DIM;
  const v16h a0 = a_frag_h16row(wrow,  0, hi);
  const v16h a1 = a_frag_h16row(wrow, 32, hi);
  const v16h a2 = a_frag_h16row(wrow, 64, hi);
  const v16h a3 = a_frag_h16row(wrow, 96, hi);

#pragma unroll
  for (int nt = 0; nt < 4; ++nt) {
    const int n = n0 + nt * 16 + l;
    const _Float16* arow = ao + n * C_DIM + hi * 16;
    v8f acc = {};
    acc = wmma16(a0, frag16(arow +  0, arow +  8), acc);
    acc = wmma16(a1, frag16(arow + 32, arow + 40), acc);
    acc = wmma16(a2, frag16(arow + 64, arow + 72), acc);
    acc = wmma16(a3, frag16(arow + 96, arow + 104), acc);
#pragma unroll
    for (int r = 0; r < 8; ++r) {
      const int o = w * 16 + r + hi * 8;
      out[o * N_DIM + n] = x[o * N_DIM + n] + acc[r] + pb[o];
    }
  }
}

// ---------------------------------------------------------------------------
extern "C" void kernel_launch(void* const* d_in, const int* in_sizes, int n_in,
                              void* d_out, int out_size, void* d_ws, size_t ws_size,
                              hipStream_t stream) {
  (void)in_sizes; (void)n_in; (void)out_size; (void)ws_size;
  const float* x    = (const float*)d_in[0];
  const float* gn_w = (const float*)d_in[1];
  const float* gn_b = (const float*)d_in[2];
  const float* qw   = (const float*)d_in[3];
  const float* qb   = (const float*)d_in[4];
  const float* kw   = (const float*)d_in[5];
  const float* kb   = (const float*)d_in[6];
  const float* vw   = (const float*)d_in[7];
  const float* vb   = (const float*)d_in[8];
  const float* pw   = (const float*)d_in[9];
  const float* pb   = (const float*)d_in[10];

  char* ws = (char*)d_ws;
  float*    stats = (float*)ws;                       // 2 floats (+pad)
  _Float16* qbuf  = (_Float16*)(ws + 256);            // 1 MB
  _Float16* kbuf  = qbuf + HEADS * HSTRIDE;           // 1 MB
  _Float16* vtbuf = kbuf + HEADS * HSTRIDE;           // 1 MB
  _Float16* aobuf = vtbuf + HEADS * HSTRIDE;          // 1 MB
  _Float16* w16   = aobuf + NTOT;                     // 4 x 32 KB f16 weights

  wconv_kernel<<<dim3(WSZ / 4 / 256, 4), 256, 0, stream>>>(qw, kw, vw, pw, w16);
  gn_stats_kernel<<<1, 1024, 0, stream>>>(x, stats);
  qkv_kernel<<<N_DIM / 64, 256, 0, stream>>>(x, gn_w, gn_b, w16, qb, kb, vb,
                                             stats, qbuf, kbuf, vtbuf);
  attn_kernel<<<(HEADS * 256) / 8, 256, 0, stream>>>(qbuf, kbuf, vtbuf, aobuf);
  proj_kernel<<<N_DIM / 64, 256, 0, stream>>>(x, w16 + 3 * WSZ, pb, aobuf,
                                              (float*)d_out);
}
